// Attention_Rel_Scl_20469814132992
// MI455X (gfx1250) — compile-verified, matrix operand order checked by reading, not run on hardware
//
#include <hip/hip_runtime.h>
#include <hip/hip_bf16.h>

// Problem constants (B,S,E,H from the reference)
#define B_ 8
#define S_ 1024
#define E_ 1024
#define H_ 16
#define D_ 64
static constexpr float kScale = 0.03125f;   // E^-0.5
static constexpr float kEps   = 1e-3f;

typedef __attribute__((ext_vector_type(16))) __bf16 v16bf;
typedef __attribute__((ext_vector_type(8)))  float  v8f;

// ---------- bf16 helpers (bit-level) ----------
__device__ __forceinline__ unsigned short f2bf_bits(float f) {
  union { float f; unsigned int u; } cv; cv.f = f;
  unsigned int u = cv.u;
  return (unsigned short)((u + 0x7FFFu + ((u >> 16) & 1u)) >> 16);  // RNE
}

// ---------- Tensor Data Mover (CDNA5 TDM) ----------
#if __has_builtin(__builtin_amdgcn_tensor_load_to_lds) && \
    __has_builtin(__builtin_amdgcn_s_wait_tensorcnt)
#define HAVE_TDM 1
typedef unsigned int u32x4 __attribute__((ext_vector_type(4)));
typedef int i32x8 __attribute__((ext_vector_type(8)));
typedef int i32x4 __attribute__((ext_vector_type(4)));

// 2D tile (elements are 2 bytes): D# per ISA 8.3/8.4; groups 2/3 NULL (<=2D tensor)
__device__ __forceinline__ void tdm_load_2d(unsigned lds_addr,
                                            unsigned long long gaddr,
                                            unsigned tile_d0, unsigned tile_d1,
                                            unsigned tensor_d0, unsigned tensor_d1,
                                            unsigned stride0) {
  u32x4 g0;
  g0[0] = 1u;                                   // count=1 (valid user D#)
  g0[1] = lds_addr;                             // lds_addr (bytes), bits 63:32
  g0[2] = (unsigned)(gaddr & 0xFFFFFFFFu);      // global_addr[31:0]
  g0[3] = (unsigned)((gaddr >> 32) & 0x01FFFFFFu) | 0x80000000u;  // [56:32]|type=2
  i32x8 g1;
  g1[0] = 0x00010000;                           // wg_mask=0, data_size=1 (2 bytes)
  g1[1] = (int)((tensor_d0 & 0xFFFFu) << 16);   // tensor_dim0 -> bits 79:48
  g1[2] = (int)(((tensor_d0 >> 16) & 0xFFFFu) | ((tensor_d1 & 0xFFFFu) << 16));
  g1[3] = (int)(((tensor_d1 >> 16) & 0xFFFFu) | ((tile_d0 & 0xFFFFu) << 16));
  g1[4] = (int)(tile_d1 & 0xFFFFu);             // tile_dim1; tile_dim2=0
  g1[5] = (int)stride0;                         // tensor_dim0_stride[31:0]
  g1[6] = 0;                                    // stride0[47:32], stride1 lo
  g1[7] = 0;
  i32x4 z4 = {0, 0, 0, 0};
#if defined(__clang_major__) && __clang_major__ >= 23
  // clang-23 / therock-10.0 headers: 6-arg form (extra int32x8 group before cpol)
  i32x8 z8 = {0, 0, 0, 0, 0, 0, 0, 0};
  __builtin_amdgcn_tensor_load_to_lds(g0, g1, z4, z4, z8, 0);
#else
  // ROCm 7.2 / clang-22: 5-arg form
  __builtin_amdgcn_tensor_load_to_lds(g0, g1, z4, z4, 0);
#endif
}
__device__ __forceinline__ unsigned lds_off(const void* p) {
  return (unsigned)(unsigned long long)(uintptr_t)p;  // LDS aperture: low 32 = offset
}
#endif

// ---------- WMMA fragment loader (CDNA5 16-bit A/B layout, wave32) ----------
// Lane L's 16 fragment elements sit at contraction offsets {kh..kh+7, 16+kh..16+kh+7}
// (kh = (L>>4)*8). With the contraction axis contiguous in LDS, a fragment is exactly
// two 16-byte runs -> two ds_load_b128, no per-element VALU assembly.
union FragCast { uint4 q[2]; v16bf bf; };

__device__ __forceinline__ v16bf load_frag(const unsigned short* __restrict__ base,
                                           int row, int ld, int col0, int lane) {
  const int kh = (lane >> 4) * 8;
  const unsigned short* rp = base + row * ld + col0 + kh;
  FragCast fc;
  fc.q[0] = *(const uint4*)(rp);
  fc.q[1] = *(const uint4*)(rp + 16);
  return fc.bf;
}

__device__ __forceinline__ v8f wmma_bf16(v16bf a, v16bf b, v8f c) {
  return __builtin_amdgcn_wmma_f32_16x16x32_bf16(false, a, false, b, (short)0, c,
                                                 false, false);
}

// ---------- Kernel 0: f32 -> bf16 convert (b128 loads, b64 stores) ----------
__global__ __launch_bounds__(256) void cvt4_kernel(const float4* __restrict__ s,
                                                   ushort4* __restrict__ d, int n4) {
  int i = blockIdx.x * blockDim.x + threadIdx.x;
  if (i < n4) {
    float4 f = s[i];
    ushort4 o;
    o.x = f2bf_bits(f.x); o.y = f2bf_bits(f.y);
    o.z = f2bf_bits(f.z); o.w = f2bf_bits(f.w);
    d[i] = o;
  }
}

// ---------- Kernel 1: QKV projection GEMM, out stored as [B,H,S,D] bf16 ----------
// WG = 256 threads (8 waves). Tile: 128(M) x 64(N), K-step 32 (= one WMMA depth).
// X tile staged via TDM (128x32 2D descriptor); W tile staged TRANSPOSED manually.
__global__ __launch_bounds__(256) void qkv_gemm_kernel(
    const unsigned short* __restrict__ Xb,   // 8192 x 1024 bf16
    const unsigned short* __restrict__ Wb,   // 1024 x 1024 bf16
    unsigned short* __restrict__ Out) {      // [B,H,S,D] bf16
  __shared__ __align__(16) unsigned short Xs[128 * 32];   // [m][k]
  __shared__ __align__(16) unsigned short WsT[64 * 32];   // [n][k]  (transposed)
  const int tid = threadIdx.x, lane = tid & 31, w = tid >> 5;
  const int m0 = blockIdx.x * 128;
  const int n0 = blockIdx.y * 64;
  const int n15 = lane & 15;

  v8f acc[4];
#pragma unroll
  for (int t = 0; t < 4; ++t)
#pragma unroll
    for (int i = 0; i < 8; ++i) acc[t][i] = 0.f;

  for (int k0 = 0; k0 < E_; k0 += 32) {
#if HAVE_TDM
    // X tile: one TDM descriptor, 128 rows x 32 cols, row stride E_ (DMA -> LDS)
    if (w == 0) {
      tdm_load_2d(lds_off(Xs),
                  (unsigned long long)(uintptr_t)(Xb + (size_t)m0 * E_ + k0),
                  /*tile*/ 32, 128, /*tensor*/ 32, 128, /*stride0*/ E_);
    }
#else
    // X tile: 128x32 = 512 uint4 (2 per thread), contraction contiguous
#pragma unroll
    for (int t = tid; t < 512; t += 256) {
      const int r = t >> 2, c4 = t & 3;
      ((uint4*)Xs)[t] = ((const uint4*)(Xb + (size_t)(m0 + r) * E_ + k0))[c4];
    }
#endif
    // W tile 32x64, stored transposed: coalesced b128 global load, scatter to [n][k]
    {
      const int r = tid >> 3, c4 = tid & 7;  // r = k-row, c4 = 8-wide n chunk
      uint4 wv = ((const uint4*)(Wb + (size_t)(k0 + r) * E_ + n0))[c4];
      const unsigned short* wsv = (const unsigned short*)&wv;
#pragma unroll
      for (int e = 0; e < 8; ++e) WsT[(c4 * 8 + e) * 32 + r] = wsv[e];
    }
#if HAVE_TDM
    if (w == 0) __builtin_amdgcn_s_wait_tensorcnt(0);
#endif
    __syncthreads();

    v16bf a = load_frag(Xs, w * 16 + n15, 32, 0, lane);   // A: rows of x strip
#pragma unroll
    for (int nt = 0; nt < 4; ++nt) {
      v16bf b = load_frag(WsT, nt * 16 + n15, 32, 0, lane);  // B: W^T rows
      acc[nt] = wmma_bf16(a, b, acc[nt]);
    }
    __syncthreads();
  }

  // C layout: VGPR r -> row r (lanes 0-15) / r+8 (lanes 16-31); col = lane&15
  const int half = lane >> 4;
#pragma unroll
  for (int nt = 0; nt < 4; ++nt)
#pragma unroll
    for (int r = 0; r < 8; ++r) {
      const int m = m0 + w * 16 + r + 8 * half;
      const int nc = n0 + nt * 16 + n15;
      const int b = m >> 10, s = m & 1023;
      const int h = nc >> 6, d = nc & 63;
      Out[(((size_t)(b * H_ + h)) * S_ + s) * D_ + d] = f2bf_bits(acc[nt][r]);
    }
}

// ---------- Kernel 2: fused flash attention + post-softmax relative bias ----------
// Grid: (S/128, H, B). WG = 256 threads (8 waves), 128 q-rows per WG (16 per wave).
// Q block and K tiles staged via TDM (contiguous -> 1D descriptors). V tile staged
// transposed [d][key] manually (PV contraction = key). Fragments = b128 LDS loads.
__global__ __launch_bounds__(256) void attn_kernel(
    const unsigned short* __restrict__ Qb, const unsigned short* __restrict__ Kb,
    const unsigned short* __restrict__ Vb, const float* __restrict__ rel_table,
    float* __restrict__ O /* [B,S,E] f32, pre-LayerNorm */) {
  __shared__ __align__(16) unsigned short Qs[128 * 64];    // [q][d]
  __shared__ __align__(16) unsigned short Ks[64 * 64];     // [key][d]
  __shared__ __align__(16) unsigned short VsT[64 * 64];    // [d][key] (transposed)
  __shared__ __align__(16) unsigned short Ps[8][16 * 64];  // per-wave P   [q][key]
  __shared__ __align__(16) unsigned short Bs[8][16 * 64];  // per-wave bias[q][key]

  const int tid = threadIdx.x, lane = tid & 31, w = tid >> 5;
  const int q0 = blockIdx.x * 128;
  const int h  = blockIdx.y;
  const int b  = blockIdx.z;
  const int half = lane >> 4, n15 = lane & 15;

  const unsigned short* Qg = Qb + (((size_t)(b * H_ + h)) * S_ + q0) * D_;
  const unsigned short* Kg = Kb + ((size_t)(b * H_ + h)) * S_ * D_;
  const unsigned short* Vg = Vb + ((size_t)(b * H_ + h)) * S_ * D_;

#if HAVE_TDM
  // Q block 128x64: rows contiguous in global -> 1D descriptor (8192 elements)
  if (w == 0) {
    tdm_load_2d(lds_off(Qs), (unsigned long long)(uintptr_t)Qg,
                /*tile*/ 8192, 1, /*tensor*/ 8192, 1, /*stride0*/ 8192);
  }
#else
#pragma unroll
  for (int t = tid; t < 1024; t += 256) ((uint4*)Qs)[t] = ((const uint4*)Qg)[t];
#endif

  v8f acc_o[4], acc_b[4];
#pragma unroll
  for (int t = 0; t < 4; ++t)
#pragma unroll
    for (int i = 0; i < 8; ++i) { acc_o[t][i] = 0.f; acc_b[t][i] = 0.f; }
  float m_i[8], l_i[8];
#pragma unroll
  for (int r = 0; r < 8; ++r) { m_i[r] = -3.0e38f; l_i[r] = 0.f; }

#if HAVE_TDM
  if (w == 0) __builtin_amdgcn_s_wait_tensorcnt(0);
#endif
  __syncthreads();

  for (int kk0 = 0; kk0 < S_; kk0 += 64) {
#if HAVE_TDM
    // K tile 64x64 contiguous -> 1D descriptor (4096 elements)
    if (w == 0) {
      tdm_load_2d(lds_off(Ks),
                  (unsigned long long)(uintptr_t)(Kg + (size_t)kk0 * D_),
                  /*tile*/ 4096, 1, /*tensor*/ 4096, 1, /*stride0*/ 4096);
    }
#else
#pragma unroll
    for (int t = tid; t < 512; t += 256)
      ((uint4*)Ks)[t] = ((const uint4*)(Kg + (size_t)kk0 * D_))[t];
#endif
    // V tile transposed into [d][key]: coalesced b128 load + 8 b16 scatters
#pragma unroll
    for (int t = tid; t < 512; t += 256) {
      const int key = t >> 3, c4 = t & 7;
      uint4 vv = ((const uint4*)(Vg + (size_t)(kk0 + key) * D_))[c4];
      const unsigned short* vsv = (const unsigned short*)&vv;
#pragma unroll
      for (int e = 0; e < 8; ++e) VsT[(c4 * 8 + e) * 64 + key] = vsv[e];
    }
#if HAVE_TDM
    if (w == 0) __builtin_amdgcn_s_wait_tensorcnt(0);
#endif
    __syncthreads();

    // S tile (16 x 64 per wave) = Q K^T : contraction over d (2 x K=32)
    v8f sacc[4];
#pragma unroll
    for (int t = 0; t < 4; ++t)
#pragma unroll
      for (int i = 0; i < 8; ++i) sacc[t][i] = 0.f;
#pragma unroll
    for (int kc = 0; kc < 2; ++kc) {
      v16bf a = load_frag(Qs, w * 16 + n15, 64, kc * 32, lane);
#pragma unroll
      for (int nt = 0; nt < 4; ++nt) {
        v16bf bk = load_frag(Ks, nt * 16 + n15, 64, kc * 32, lane);
        sacc[nt] = wmma_bf16(a, bk, sacc[nt]);
      }
    }

    // Online softmax update (rows live across the 16-lane half-groups)
#pragma unroll
    for (int r = 0; r < 8; ++r) {
#pragma unroll
      for (int nt = 0; nt < 4; ++nt) sacc[nt][r] *= kScale;
      float tmax = fmaxf(fmaxf(sacc[0][r], sacc[1][r]),
                         fmaxf(sacc[2][r], sacc[3][r]));
#pragma unroll
      for (int mask = 1; mask < 16; mask <<= 1)
        tmax = fmaxf(tmax, __shfl_xor(tmax, mask, 32));
      const float mnew  = fmaxf(m_i[r], tmax);
      const float alpha = __expf(m_i[r] - mnew);
      m_i[r] = mnew;
      float rsum = 0.f;
#pragma unroll
      for (int nt = 0; nt < 4; ++nt) {
        const float p = __expf(sacc[nt][r] - mnew);
        rsum += p;
        Ps[w][(r + 8 * half) * 64 + nt * 16 + n15] = f2bf_bits(p);
        acc_o[nt][r] *= alpha;
      }
#pragma unroll
      for (int mask = 1; mask < 16; mask <<= 1)
        rsum += __shfl_xor(rsum, mask, 32);
      l_i[r] = l_i[r] * alpha + rsum;
    }

    // Relative bias tile via closed-form index of the quirky flat reshape:
    // bias[h,q,k] = rel_table[clip((q&15)*64+(k>>4) - (h*64+(q>>4)) + 1023), k&15]
    for (int t = lane; t < 16 * 64; t += 32) {
      const int row = t >> 6, kc = t & 63;
      const int q = q0 + w * 16 + row;
      const int k = kk0 + kc;
      const int hh = k & 15;
      const int j  = (q & 15) * 64 + (k >> 4);
      const int i2 = h * 64 + (q >> 4);
      int idx = j - i2 + (S_ - 1);
      idx = idx < 0 ? 0 : (idx > 2 * S_ - 2 ? 2 * S_ - 2 : idx);
      Bs[w][t] = f2bf_bits(rel_table[idx * H_ + hh]);
    }

    // acc_o += P @ V ; acc_b += bias @ V  (contraction over 64 keys, contiguous)
#pragma unroll
    for (int kc = 0; kc < 2; ++kc) {
      v16bf ap = load_frag(Ps[w], n15, 64, kc * 32, lane);
      v16bf ab = load_frag(Bs[w], n15, 64, kc * 32, lane);
#pragma unroll
      for (int nt = 0; nt < 4; ++nt) {
        v16bf bv = load_frag(VsT, nt * 16 + n15, 64, kc * 32, lane);
        acc_o[nt] = wmma_bf16(ap, bv, acc_o[nt]);
        acc_b[nt] = wmma_bf16(ab, bv, acc_b[nt]);
      }
    }
    __syncthreads();  // protect Ks/VsT before next tile's loads
  }

  // out = softmax part / l + bias part ; write merged-head layout [b, q, h*64+d]
#pragma unroll
  for (int nt = 0; nt < 4; ++nt)
#pragma unroll
    for (int r = 0; r < 8; ++r) {
      const int q = q0 + w * 16 + r + 8 * half;
      const int d = nt * 16 + n15;
      const float val = acc_o[nt][r] / l_i[r] + acc_b[nt][r];
      O[((size_t)b * S_ + q) * E_ + h * D_ + d] = val;
    }
}

// ---------- Kernel 3: LayerNorm over last axis ----------
__global__ __launch_bounds__(256) void ln_kernel(const float* __restrict__ X,
                                                 const float* __restrict__ gamma,
                                                 const float* __restrict__ beta,
                                                 float* __restrict__ Y) {
  __shared__ float2 red[256];
  const int row = blockIdx.x;
  const int tid = threadIdx.x;
  const float* xr = X + (size_t)row * E_;
  float v[4], s = 0.f, sq = 0.f;
#pragma unroll
  for (int i = 0; i < 4; ++i) {
    v[i] = xr[tid + i * 256];
    s += v[i];
    sq += v[i] * v[i];
  }
  red[tid] = make_float2(s, sq);
  __syncthreads();
  for (int st = 128; st > 0; st >>= 1) {
    if (tid < st) {
      red[tid].x += red[tid + st].x;
      red[tid].y += red[tid + st].y;
    }
    __syncthreads();
  }
  const float mean = red[0].x * (1.f / E_);
  const float var  = red[0].y * (1.f / E_) - mean * mean;
  const float rstd = rsqrtf(var + kEps);
#pragma unroll
  for (int i = 0; i < 4; ++i) {
    const int c = tid + i * 256;
    Y[(size_t)row * E_ + c] = (v[i] - mean) * rstd * gamma[c] + beta[c];
  }
}

// ---------- Launch ----------
extern "C" void kernel_launch(void* const* d_in, const int* in_sizes, int n_in,
                              void* d_out, int out_size, void* d_ws, size_t ws_size,
                              hipStream_t stream) {
  (void)in_sizes; (void)n_in; (void)out_size; (void)ws_size;
  const float* x   = (const float*)d_in[0];
  const float* Wq  = (const float*)d_in[1];
  const float* Wk  = (const float*)d_in[2];
  const float* Wv  = (const float*)d_in[3];
  const float* rel = (const float*)d_in[4];
  const float* gamma = (const float*)d_in[5];
  const float* beta  = (const float*)d_in[6];

  // Workspace layout (~107 MB total)
  unsigned short* xb  = (unsigned short*)d_ws;          // 8192x1024 bf16
  unsigned short* wqb = xb + (size_t)8192 * 1024;
  unsigned short* wkb = wqb + (size_t)1024 * 1024;
  unsigned short* wvb = wkb + (size_t)1024 * 1024;
  unsigned short* Qb  = wvb + (size_t)1024 * 1024;      // [B,H,S,D] bf16
  unsigned short* Kb  = Qb + (size_t)8192 * 1024;
  unsigned short* Vb  = Kb + (size_t)8192 * 1024;
  float* O = (float*)(Vb + (size_t)8192 * 1024);        // [B,S,E] f32

  // Converts
  {
    const int n4x = (B_ * S_ * E_) / 4;                 // 2,097,152
    cvt4_kernel<<<(n4x + 255) / 256, 256, 0, stream>>>((const float4*)x,
                                                       (ushort4*)xb, n4x);
    const int n4w = (E_ * E_) / 4;                      // 262,144
    cvt4_kernel<<<(n4w + 255) / 256, 256, 0, stream>>>((const float4*)Wq,
                                                       (ushort4*)wqb, n4w);
    cvt4_kernel<<<(n4w + 255) / 256, 256, 0, stream>>>((const float4*)Wk,
                                                       (ushort4*)wkb, n4w);
    cvt4_kernel<<<(n4w + 255) / 256, 256, 0, stream>>>((const float4*)Wv,
                                                       (ushort4*)wvb, n4w);
  }

  // QKV projections: M=8192, N=1024, K=1024 each
  dim3 ggrid(8192 / 128, 1024 / 64);
  qkv_gemm_kernel<<<ggrid, 256, 0, stream>>>(xb, wqb, Qb);
  qkv_gemm_kernel<<<ggrid, 256, 0, stream>>>(xb, wkb, Kb);
  qkv_gemm_kernel<<<ggrid, 256, 0, stream>>>(xb, wvb, Vb);

  // Fused flash attention + relative bias
  attn_kernel<<<dim3(S_ / 128, H_, B_), 256, 0, stream>>>(Qb, Kb, Vb, rel, O);

  // LayerNorm epilogue
  ln_kernel<<<B_ * S_, 256, 0, stream>>>(O, gamma, beta, (float*)d_out);
}
